// SGLangTPFusedMoEAdapter_79783312491223
// MI455X (gfx1250) — matrix operand order repaired and checked by reading, not compiled, not run on hardware
//
#include <hip/hip_runtime.h>

// Problem constants (match reference).
#define T_TOK 1024
#define H_DIM 2048
#define I_DIM 5632
#define E_NUM 8
#define K_TOP 2
#define P_MAX (T_TOK * K_TOP)   // 2048 routed (token, expert) pairs

typedef __attribute__((ext_vector_type(16))) __bf16 v16bf;
typedef __attribute__((ext_vector_type(8)))  __bf16 bf16x8;
typedef __attribute__((ext_vector_type(4)))  __bf16 bf16x4;
typedef __attribute__((ext_vector_type(8)))  float  v8f;
typedef __attribute__((ext_vector_type(4)))  int    v4i;

union Frag { v16bf v; bf16x8 h[2]; };

#define AP 72   // LDS row pitch in bf16 (BK=64 + 8 pad; 144B rows, 16B aligned)
#define BP 72

#if __has_builtin(__builtin_amdgcn_global_load_async_to_lds_b128) && \
    __has_builtin(__builtin_amdgcn_s_wait_asynccnt)
#  define USE_ASYNC_COPY 1
typedef __attribute__((address_space(1))) v4i* gptr_v4i;
typedef __attribute__((address_space(3))) v4i* lptr_v4i;
#else
#  define USE_ASYNC_COPY 0
#endif

__device__ __forceinline__ float fast_silu(float g) {
    // g * rcp(1 + exp(-g)) : one v_exp + one v_rcp instead of IEEE div ladder
    return g * __builtin_amdgcn_rcpf(1.0f + __expf(-g));
}

// ---------------------------------------------------------------------------
// Kernel 0: zero output + per-expert counters
// ---------------------------------------------------------------------------
__global__ void moe_zero_kernel(float* __restrict__ out, int n, int* __restrict__ cnt) {
    int i = blockIdx.x * blockDim.x + threadIdx.x;
    if (i < n) out[i] = 0.0f;
    if (blockIdx.x == 0 && threadIdx.x < E_NUM) cnt[threadIdx.x] = 0;
}

// ---------------------------------------------------------------------------
// Kernel 1: route pairs into per-expert lists
// ---------------------------------------------------------------------------
__global__ void moe_route_kernel(const int* __restrict__ topk_ids,
                                 int* __restrict__ cnt,
                                 int* __restrict__ lists) {
    int p = blockIdx.x * blockDim.x + threadIdx.x;
    if (p >= P_MAX) return;
    int e = topk_ids[p];
    int slot = atomicAdd(&cnt[e], 1);
    lists[e * P_MAX + slot] = p;
}

// ---------------------------------------------------------------------------
// Kernel 2: grouped GEMM13 + SwiGLU -> bf16 activations
// Tile: BM=64 tokens x BN=128 I-cols, BK=64 over H (two 16x16x32 WMMA steps
// per LDS slab -> 16 WMMAs between barriers). 8 waves in a 2x4 grid.
// ---------------------------------------------------------------------------
__global__ __launch_bounds__(256)
void moe_gemm13_kernel(const float* __restrict__ hidden,
                       const float* __restrict__ w13,
                       const int*   __restrict__ cnt,
                       const int*   __restrict__ lists,
                       __bf16*      __restrict__ act) {
    const int e  = blockIdx.z;
    const int m0 = blockIdx.y * 64;
    const int i0 = blockIdx.x * 128;
    const int cnt_e = cnt[e];
    if (m0 >= cnt_e) return;   // uniform early-exit for over-provisioned grid

    __shared__ __align__(16) __bf16 As[64 * AP];
    __shared__ __align__(16) __bf16 Bg[128 * BP];
    __shared__ __align__(16) __bf16 Bu[128 * BP];
    __shared__ int sP[64];

    const int tid  = threadIdx.x;
    const int lane = tid & 31;
    const int wave = tid >> 5;
    const int wr = wave >> 2;     // 0..1 : M wave-row
    const int wc = wave & 3;      // 0..3 : N wave-col

    if (tid < 64) {
        int r = m0 + tid;
        sP[tid] = (r < cnt_e) ? lists[e * P_MAX + r] : -1;
    }
    __syncthreads();

    const size_t w13_gate = (size_t)e * (2 * (size_t)I_DIM) * H_DIM + (size_t)i0 * H_DIM;
    const size_t w13_up   = w13_gate + (size_t)I_DIM * H_DIM;

    v8f acc_g[2][2] = {};
    v8f acc_u[2][2] = {};

    const int lm  = lane & 15;
    const int klo = (lane < 16) ? 0 : 16;  // 16-bit A frag K split (contiguous, matches B)
    const int khi = (lane < 16) ? 0 : 16;  // 16-bit B frag K split

    for (int kt = 0; kt < H_DIM / 64; ++kt) {
        const int kk = kt * 64;
        // stage A: 64 rows x 64 k fp32 gathered -> bf16 LDS (1024 float4)
#pragma unroll
        for (int j = 0; j < 4; ++j) {
            int idx = tid + j * 256;           // 0..1023
            int r = idx >> 4;
            int c = (idx & 15) * 4;
            int p = sP[r];
            int t = (p < 0) ? 0 : (p >> 1);    // K_TOP == 2
            float4 f = *(const float4*)(hidden + (size_t)t * H_DIM + kk + c);
            bf16x4 b; b.x = (__bf16)f.x; b.y = (__bf16)f.y; b.z = (__bf16)f.z; b.w = (__bf16)f.w;
            *(bf16x4*)(&As[r * AP + c]) = b;
        }
        // stage B gate + up: 128 rows x 64 k fp32 -> bf16 LDS (2048 float4 each)
#pragma unroll
        for (int j = 0; j < 8; ++j) {
            int idx = tid + j * 256;           // 0..2047
            int n = idx >> 4;
            int c = (idx & 15) * 4;
            float4 fg = *(const float4*)(w13 + w13_gate + (size_t)n * H_DIM + kk + c);
            float4 fu = *(const float4*)(w13 + w13_up   + (size_t)n * H_DIM + kk + c);
            bf16x4 bg; bg.x=(__bf16)fg.x; bg.y=(__bf16)fg.y; bg.z=(__bf16)fg.z; bg.w=(__bf16)fg.w;
            bf16x4 bu; bu.x=(__bf16)fu.x; bu.y=(__bf16)fu.y; bu.z=(__bf16)fu.z; bu.w=(__bf16)fu.w;
            *(bf16x4*)(&Bg[n * BP + c]) = bg;
            *(bf16x4*)(&Bu[n * BP + c]) = bu;
        }
        __syncthreads();

#pragma unroll
        for (int kc = 0; kc < 2; ++kc) {       // two 32-deep WMMA steps per slab
            const int kb = kc * 32;
            Frag a[2];
#pragma unroll
            for (int ti = 0; ti < 2; ++ti) {
                const __bf16* ap = &As[(wr * 32 + ti * 16 + lm) * AP + kb];
                a[ti].h[0] = *(const bf16x8*)(ap + klo);
                a[ti].h[1] = *(const bf16x8*)(ap + klo + 8);
            }
#pragma unroll
            for (int tj = 0; tj < 2; ++tj) {
                const int n = wc * 32 + tj * 16 + lm;
                Frag bg, bu;
                bg.h[0] = *(const bf16x8*)(&Bg[n * BP + kb + khi]);
                bg.h[1] = *(const bf16x8*)(&Bg[n * BP + kb + khi + 8]);
                bu.h[0] = *(const bf16x8*)(&Bu[n * BP + kb + khi]);
                bu.h[1] = *(const bf16x8*)(&Bu[n * BP + kb + khi + 8]);
#pragma unroll
                for (int ti = 0; ti < 2; ++ti) {
                    acc_g[ti][tj] = __builtin_amdgcn_wmma_f32_16x16x32_bf16(
                        false, a[ti].v, false, bg.v, (short)0, acc_g[ti][tj], false, false);
                    acc_u[ti][tj] = __builtin_amdgcn_wmma_f32_16x16x32_bf16(
                        false, a[ti].v, false, bu.v, (short)0, acc_u[ti][tj], false, false);
                }
            }
        }
        __syncthreads();
    }

    // epilogue: SwiGLU, store bf16 activation rows
    const int mhalf = (lane < 16) ? 0 : 8;
#pragma unroll
    for (int ti = 0; ti < 2; ++ti)
#pragma unroll
        for (int tj = 0; tj < 2; ++tj)
#pragma unroll
            for (int r = 0; r < 8; ++r) {
                int row = wr * 32 + ti * 16 + mhalf + r;
                int p = sP[row];
                if (p < 0) continue;
                int n = i0 + wc * 32 + tj * 16 + lm;
                float u = acc_u[ti][tj][r];
                act[(size_t)p * I_DIM + n] = (__bf16)(fast_silu(acc_g[ti][tj][r]) * u);
            }
}

// ---------------------------------------------------------------------------
// Kernel 3: grouped GEMM2 + weighted atomic combine
//   out[t, h] += topk_w[p] * (act[p] . w2[e, h, :])
// A (bf16 act) staged to LDS via GLOBAL_LOAD_ASYNC_TO_LDS_B128 when available
// (pure byte copy -> ASYNCcnt path); w2 staged fp32->bf16 through registers.
// ---------------------------------------------------------------------------
__global__ __launch_bounds__(256)
void moe_gemm2_kernel(const __bf16* __restrict__ act,
                      const float*  __restrict__ w2,
                      const float*  __restrict__ topk_w,
                      const int*    __restrict__ cnt,
                      const int*    __restrict__ lists,
                      float*        __restrict__ out) {
    const int e  = blockIdx.z;
    const int m0 = blockIdx.y * 64;
    const int h0 = blockIdx.x * 128;
    const int cnt_e = cnt[e];
    if (m0 >= cnt_e) return;

    __shared__ __align__(16) __bf16 As2[64 * AP];
    __shared__ __align__(16) __bf16 Bs[128 * BP];
    __shared__ int   sP[64];
    __shared__ float sW[64];

    const int tid  = threadIdx.x;
    const int lane = tid & 31;
    const int wave = tid >> 5;
    const int wr = wave >> 2, wc = wave & 3;

    if (tid < 64) {
        int r = m0 + tid;
        int p = (r < cnt_e) ? lists[e * P_MAX + r] : -1;
        sP[tid] = p;
        sW[tid] = (p < 0) ? 0.0f : topk_w[p];
    }
    __syncthreads();

    const int lm  = lane & 15;
    const int klo = (lane < 16) ? 0 : 16;
    const int khi = (lane < 16) ? 0 : 16;

    // Per-thread A-copy assignment: 64 rows x 64 k bf16 = 512 16B chunks,
    // 2 chunks/thread: row r = idx>>3, chunk c = idx&7 (16B at k = c*8).
    const __bf16* aSrc[2];
    unsigned      aDst[2];
#pragma unroll
    for (int j = 0; j < 2; ++j) {
        int idx = tid + j * 256;
        int r = idx >> 3;
        int c = idx & 7;
        int p = sP[r];
        aSrc[j] = act + (size_t)((p < 0) ? 0 : p) * I_DIM + c * 8;
        aDst[j] = r * AP + c * 8;
    }

    const size_t w2_base = (size_t)e * H_DIM * I_DIM + (size_t)h0 * I_DIM;

    v8f acc[2][2] = {};

    for (int kt = 0; kt < I_DIM / 64; ++kt) {
        const int kk = kt * 64;
        // ---- stage A tile (bf16 copy): async DMA to LDS when available
#pragma unroll
        for (int j = 0; j < 2; ++j) {
#if USE_ASYNC_COPY
            __builtin_amdgcn_global_load_async_to_lds_b128(
                (gptr_v4i)(void*)const_cast<__bf16*>(aSrc[j] + kk),
                (lptr_v4i)(void*)&As2[aDst[j]],
                0, 0);
#else
            *(bf16x8*)(&As2[aDst[j]]) = *(const bf16x8*)(aSrc[j] + kk);
#endif
        }
        // ---- stage B tile: 128 rows x 64 k fp32 -> bf16 LDS
#pragma unroll
        for (int j = 0; j < 8; ++j) {
            int idx = tid + j * 256;
            int n = idx >> 4;
            int c = (idx & 15) * 4;
            float4 f = *(const float4*)(w2 + w2_base + (size_t)n * I_DIM + kk + c);
            bf16x4 b; b.x=(__bf16)f.x; b.y=(__bf16)f.y; b.z=(__bf16)f.z; b.w=(__bf16)f.w;
            *(bf16x4*)(&Bs[n * BP + c]) = b;
        }
#if USE_ASYNC_COPY
        __builtin_amdgcn_s_wait_asynccnt(0);
#endif
        __syncthreads();

#pragma unroll
        for (int kc = 0; kc < 2; ++kc) {
            const int kb = kc * 32;
            Frag a[2];
#pragma unroll
            for (int ti = 0; ti < 2; ++ti) {
                const __bf16* ap = &As2[(wr * 32 + ti * 16 + lm) * AP + kb];
                a[ti].h[0] = *(const bf16x8*)(ap + klo);
                a[ti].h[1] = *(const bf16x8*)(ap + klo + 8);
            }
#pragma unroll
            for (int tj = 0; tj < 2; ++tj) {
                const int n = wc * 32 + tj * 16 + lm;
                Frag b;
                b.h[0] = *(const bf16x8*)(&Bs[n * BP + kb + khi]);
                b.h[1] = *(const bf16x8*)(&Bs[n * BP + kb + khi + 8]);
#pragma unroll
                for (int ti = 0; ti < 2; ++ti)
                    acc[ti][tj] = __builtin_amdgcn_wmma_f32_16x16x32_bf16(
                        false, a[ti].v, false, b.v, (short)0, acc[ti][tj], false, false);
            }
        }
        __syncthreads();
    }

    const int mhalf = (lane < 16) ? 0 : 8;
#pragma unroll
    for (int ti = 0; ti < 2; ++ti)
#pragma unroll
        for (int tj = 0; tj < 2; ++tj)
#pragma unroll
            for (int r = 0; r < 8; ++r) {
                int row = wr * 32 + ti * 16 + mhalf + r;
                int p = sP[row];
                if (p < 0) continue;
                int t = p >> 1;                         // K_TOP == 2
                int h = h0 + wc * 32 + tj * 16 + lm;
                atomicAdd(&out[(size_t)t * H_DIM + h], sW[row] * acc[ti][tj][r]);
            }
}

// ---------------------------------------------------------------------------
// Launch
// ---------------------------------------------------------------------------
extern "C" void kernel_launch(void* const* d_in, const int* in_sizes, int n_in,
                              void* d_out, int out_size, void* d_ws, size_t ws_size,
                              hipStream_t stream) {
    (void)in_sizes; (void)n_in; (void)out_size; (void)ws_size;
    const float* hidden  = (const float*)d_in[0];
    const int*   topkid  = (const int*)d_in[1];
    const float* topkw   = (const float*)d_in[2];
    // d_in[3] router_logits: unused by the reference output
    const float* w13     = (const float*)d_in[4];
    const float* w2      = (const float*)d_in[5];
    float* out = (float*)d_out;

    char* ws = (char*)d_ws;
    int*    cnt   = (int*)ws;                       // [E]
    int*    lists = (int*)(ws + 256);               // [E][P_MAX]
    __bf16* act   = (__bf16*)(ws + (1 << 17));      // [P_MAX][I] bf16 (~23 MB)

    const int nOut = T_TOK * H_DIM;
    moe_zero_kernel<<<(nOut + 255) / 256, 256, 0, stream>>>(out, nOut, cnt);
    moe_route_kernel<<<(P_MAX + 255) / 256, 256, 0, stream>>>(topkid, cnt, lists);

    dim3 g13(I_DIM / 128, P_MAX / 64, E_NUM);   // (44, 32, 8)
    moe_gemm13_kernel<<<g13, 256, 0, stream>>>(hidden, w13, cnt, lists, act);

    dim3 g2(H_DIM / 128, P_MAX / 64, E_NUM);    // (16, 32, 8)
    moe_gemm2_kernel<<<g2, 256, 0, stream>>>(act, w2, topkw, cnt, lists, out);
}
